// HCTNet_Encoder_87325275062908
// MI455X (gfx1250) — compile-verified, matrix operand order checked by reading, not run on hardware
//
#include <hip/hip_runtime.h>
#include <hip/hip_bf16.h>
#include <math.h>

typedef __attribute__((ext_vector_type(16))) __bf16 v16bf;
typedef __attribute__((ext_vector_type(8)))  __bf16 v8bf;
typedef __attribute__((ext_vector_type(8)))  float  v8f;

#define BNS_CONST 0.999995000037499f  /* 1/sqrt(1+1e-5) */
#define ALD 40   /* bf16 row stride: 80B = 16B-aligned rows */

__device__ __forceinline__ float gelu_f(float x) {
  return 0.5f * x * (1.0f + erff(x * 0.70710678118654752f));
}

// 16-element bf16 fragment from two aligned 16B LDS runs.
// elems[0..7] = p[half*8 ..], elems[8..15] = p[16+half*8 ..]
__device__ __forceinline__ v16bf frag16(const __bf16* p, int half) {
  const v8bf* q = (const v8bf*)p;
  v8bf lo = q[half];
  v8bf hi = q[2 + half];
  return __builtin_shufflevector(lo, hi, 0, 1, 2, 3, 4, 5, 6, 7,
                                 8, 9, 10, 11, 12, 13, 14, 15);
}

// Same fragment pattern but source is f32 in LDS (softmax probs).
__device__ __forceinline__ v16bf frag16_f32(const float* p, int half) {
  const float4* q = (const float4*)p;
  float4 f0 = q[half * 2], f1 = q[half * 2 + 1];
  float4 f2 = q[4 + half * 2], f3 = q[5 + half * 2];
  v16bf a;
  a[0] = (__bf16)f0.x;  a[1] = (__bf16)f0.y;  a[2] = (__bf16)f0.z;  a[3] = (__bf16)f0.w;
  a[4] = (__bf16)f1.x;  a[5] = (__bf16)f1.y;  a[6] = (__bf16)f1.z;  a[7] = (__bf16)f1.w;
  a[8] = (__bf16)f2.x;  a[9] = (__bf16)f2.y;  a[10] = (__bf16)f2.z; a[11] = (__bf16)f2.w;
  a[12] = (__bf16)f3.x; a[13] = (__bf16)f3.y; a[14] = (__bf16)f3.z; a[15] = (__bf16)f3.w;
  return a;
}

union P4 { __bf16 h[4]; uint2 u; };

// ---------------------------------------------------------------------------
// Generic WMMA GEMM:  C[z] = epilogue( A[z] (MxK) * B[z] (KxN) )
//   Tile 64x128x32; 128 threads = 4 wave32 waves in a 2x2 grid; each wave
//   computes 32 rows x 64 cols = 8 WMMA accumulators (1.5 b128 LDS loads per
//   WMMA).  A staged row-major bf16 (stride 40); B staged TRANSPOSED
//   (col-major, stride 40) so B fragments are contiguous ds_load_b128 runs.
//   B path: dense row-major (ldb) or implicit im2col gather (convGather).
//   Epilogue: per-row affine (BN, folded 1/sqrt(1+eps)), per-col bias, GELU,
//   residual add.
// ---------------------------------------------------------------------------
__global__ __launch_bounds__(128)
void k_gemm(const float* __restrict__ A, long aBatch, int lda,
            const float* __restrict__ Bm, long bBatch, int ldb,
            float* __restrict__ C, long cBatch,
            int M, int N, int K,
            int convGather, int ks, int stride, int pad, int inW, int inH, int outW,
            const float* __restrict__ rowScale, const float* __restrict__ rowBias,
            float scaleMul,
            const float* __restrict__ colBias,
            const float* __restrict__ residual, long resBatch,
            int geluFlag)
{
  __shared__ __bf16 As[64 * ALD];    // [row][k]
  __shared__ __bf16 Bst[128 * ALD];  // [n][k]  (transposed)
  const int tid  = threadIdx.x;
  const int wave = tid >> 5;
  const int lane = tid & 31;
  const int half = (lane >> 4) & 1;
  const int l15  = lane & 15;
  const int wy = wave >> 1, wx = wave & 1;
  const int z = blockIdx.z;
  const int mBase = blockIdx.y * 64, nBase = blockIdx.x * 128;
  const float* Ab = A + (long)z * aBatch;
  const float* Bb = Bm + (long)z * bBatch;
  const int ksq = ks * ks;
  const bool a4 = ((lda & 3) == 0);
  const bool b4 = ((ldb & 3) == 0);

  v8f acc[2][4] = {};

  for (int kb = 0; kb < K; kb += 32) {
    // ---- stage A (64x32) -> bf16 rows, float4 reads, uint2 packed stores
    for (int idx = tid * 4; idx < 64 * 32; idx += 512) {
      int r = idx >> 5, c = idx & 31;
      int m = mBase + r;
      P4 p;
      if (m < M && a4 && (kb + c + 3) < K) {
        const float4 f = *(const float4*)(Ab + (long)m * lda + kb + c);
        p.h[0] = (__bf16)f.x; p.h[1] = (__bf16)f.y;
        p.h[2] = (__bf16)f.z; p.h[3] = (__bf16)f.w;
      } else {
#pragma unroll
        for (int j = 0; j < 4; j++) {
          int k = kb + c + j;
          float v = (m < M && k < K) ? Ab[(long)m * lda + k] : 0.f;
          p.h[j] = (__bf16)v;
        }
      }
      *(uint2*)&As[r * ALD + c] = p.u;
    }
    // ---- stage B (32x128) transposed into Bst[n][k]
    for (int idx = tid * 4; idx < 32 * 128; idx += 512) {
      int r = idx >> 7, c = idx & 127;  // r = k-offset, c = n-offset
      int k = kb + r;
      float v4[4];
      if (!convGather && b4 && k < K && (nBase + c + 3) < N) {
        const float4 f = *(const float4*)(Bb + (long)k * ldb + nBase + c);
        v4[0] = f.x; v4[1] = f.y; v4[2] = f.z; v4[3] = f.w;
      } else {
#pragma unroll
        for (int j = 0; j < 4; j++) {
          int n = nBase + c + j;
          float v = 0.f;
          if (k < K && n < N) {
            if (!convGather) {
              v = Bb[(long)k * ldb + n];
            } else {
              int ch = k / ksq, rs = k - ch * ksq;
              int rr = rs / ks, ss = rs - rr * ks;
              int py = n / outW, px = n - py * outW;
              int iy = py * stride - pad + rr, ix = px * stride - pad + ss;
              if (iy >= 0 && iy < inH && ix >= 0 && ix < inW)
                v = Bb[(long)ch * inW * inH + (long)iy * inW + ix];
            }
          }
          v4[j] = v;
        }
      }
#pragma unroll
      for (int j = 0; j < 4; j++) Bst[(c + j) * ALD + r] = (__bf16)v4[j];
    }
    __syncthreads();

    v16bf af[2], bf[4];
#pragma unroll
    for (int r2 = 0; r2 < 2; r2++)
      af[r2] = frag16(&As[(wy * 32 + r2 * 16 + l15) * ALD], half);
#pragma unroll
    for (int j = 0; j < 4; j++)
      bf[j] = frag16(&Bst[(wx * 64 + j * 16 + l15) * ALD], half);
#pragma unroll
    for (int r2 = 0; r2 < 2; r2++)
#pragma unroll
      for (int j = 0; j < 4; j++)
        acc[r2][j] = __builtin_amdgcn_wmma_f32_16x16x32_bf16(
            false, af[r2], false, bf[j], (short)0, acc[r2][j], false, false);
    __syncthreads();
  }

  float* Cb = C + (long)z * cBatch;
  const float* Rb = residual ? residual + (long)z * resBatch : nullptr;
#pragma unroll
  for (int r2 = 0; r2 < 2; r2++) {
#pragma unroll
    for (int j = 0; j < 4; j++) {
#pragma unroll
      for (int e = 0; e < 8; e++) {
        int m = mBase + wy * 32 + r2 * 16 + half * 8 + e;
        int n = nBase + wx * 64 + j * 16 + l15;
        if (m < M && n < N) {
          float v = acc[r2][j][e];
          if (rowScale) v = v * rowScale[m] * scaleMul + rowBias[m];
          if (colBias)  v += colBias[n];
          if (geluFlag) v = gelu_f(v);
          if (Rb)       v += Rb[(long)m * N + n];
          Cb[(long)m * N + n] = v;
        }
      }
    }
  }
}

// ---------------------------------------------------------------------------
// Fused multi-head attention for one (batch, head, 64-query strip).
// qkv: (B, Ntok, 768) with q|k|v each 256 = 8 heads * 32.
// out: (B, Ntok, 256).
// LDS: Qs[64][40] bf16, Ks[Ntok][40] bf16, Vst[32][Ntok+8] bf16 (transposed),
//      dots[64][Ntok] f32.
// ---------------------------------------------------------------------------
__global__ __launch_bounds__(128)
void k_attn(const float* __restrict__ qkv, const float* __restrict__ relTable,
            float* __restrict__ out, int Ntok, int ih, int iw)
{
  extern __shared__ char smem[];
  const int NV = Ntok + 8;
  __bf16* Qs  = (__bf16*)smem;           // 64 * ALD
  __bf16* Ks  = Qs + 64 * ALD;           // Ntok * ALD
  __bf16* Vst = Ks + Ntok * ALD;         // 32 * NV   [dim][key]
  float*  dots = (float*)(Vst + 32 * NV);

  const int tid  = threadIdx.x;
  const int wave = tid >> 5;
  const int lane = tid & 31;
  const int half = (lane >> 4) & 1;
  const int l15  = lane & 15;
  const int head = blockIdx.y, z = blockIdx.z;
  const int qBase = blockIdx.x * 64;
  const float* base = qkv + (long)z * Ntok * 768;

  // ---- stage Q rows
  for (int idx = tid * 4; idx < 64 * 32; idx += 512) {
    int r = idx >> 5, c = idx & 31;
    const float4 f = *(const float4*)(base + (long)(qBase + r) * 768 + head * 32 + c);
    P4 p;
    p.h[0] = (__bf16)f.x; p.h[1] = (__bf16)f.y;
    p.h[2] = (__bf16)f.z; p.h[3] = (__bf16)f.w;
    *(uint2*)&Qs[r * ALD + c] = p.u;
  }
  // ---- stage K rows + V transposed
  for (int idx = tid * 4; idx < Ntok * 32; idx += 512) {
    int r = idx >> 5, c = idx & 31;
    const float4 fk = *(const float4*)(base + (long)r * 768 + 256 + head * 32 + c);
    P4 pk;
    pk.h[0] = (__bf16)fk.x; pk.h[1] = (__bf16)fk.y;
    pk.h[2] = (__bf16)fk.z; pk.h[3] = (__bf16)fk.w;
    *(uint2*)&Ks[r * ALD + c] = pk.u;
    const float4 fv = *(const float4*)(base + (long)r * 768 + 512 + head * 32 + c);
    Vst[(c + 0) * NV + r] = (__bf16)fv.x;
    Vst[(c + 1) * NV + r] = (__bf16)fv.y;
    Vst[(c + 2) * NV + r] = (__bf16)fv.z;
    Vst[(c + 3) * NV + r] = (__bf16)fv.w;
  }
  __syncthreads();

  // ---- dots = Q * K^T * scale + rel_bias ----
  v16bf a = frag16(&Qs[(wave * 16 + l15) * ALD], half);
  const int riw = 2 * iw - 1;
  for (int kt = 0; kt < Ntok; kt += 16) {
    // B-fragment of K^T == A-pattern of K rows
    v16bf b = frag16(&Ks[(kt + l15) * ALD], half);
    v8f c = {};
    c = __builtin_amdgcn_wmma_f32_16x16x32_bf16(
        false, a, false, b, (short)0, c, false, false);
#pragma unroll
    for (int e = 0; e < 8; e++) {
      int mL = wave * 16 + half * 8 + e;
      int n  = kt + l15;
      int qi = qBase + mL, kj = n;
      int yi = qi / iw, xi = qi - yi * iw;
      int yk = kj / iw, xk = kj - yk * iw;
      int idx = (yi - yk + ih - 1) * riw + (xi - xk + iw - 1);
      dots[mL * Ntok + n] = c[e] * 0.17677669529663687f + relTable[idx * 8 + head];
    }
  }
  __syncthreads();

  // ---- row softmax (each wave owns its 16 rows) ----
  for (int rr = 0; rr < 16; rr++) {
    int row = wave * 16 + rr;
    float mx = -3.4e38f;
    for (int c = lane; c < Ntok; c += 32) mx = fmaxf(mx, dots[row * Ntok + c]);
#pragma unroll
    for (int o = 16; o > 0; o >>= 1) mx = fmaxf(mx, __shfl_xor(mx, o, 32));
    float sm = 0.f;
    for (int c = lane; c < Ntok; c += 32) {
      float e2 = __expf(dots[row * Ntok + c] - mx);
      dots[row * Ntok + c] = e2;
      sm += e2;
    }
#pragma unroll
    for (int o = 16; o > 0; o >>= 1) sm += __shfl_xor(sm, o, 32);
    float inv = 1.f / sm;
    for (int c = lane; c < Ntok; c += 32) dots[row * Ntok + c] *= inv;
  }
  __syncthreads();

  // ---- out = attn * V ----
  v8f accd[2] = {};
  for (int kc = 0; kc < Ntok; kc += 32) {
    v16bf aa = frag16_f32(&dots[(wave * 16 + l15) * Ntok + kc], half);
#pragma unroll
    for (int j = 0; j < 2; j++) {
      v16bf b = frag16(&Vst[(j * 16 + l15) * NV + kc], half);
      accd[j] = __builtin_amdgcn_wmma_f32_16x16x32_bf16(
          false, aa, false, b, (short)0, accd[j], false, false);
    }
  }
  float* ob = out + (long)z * Ntok * 256;
#pragma unroll
  for (int j = 0; j < 2; j++)
#pragma unroll
    for (int e = 0; e < 8; e++) {
      int mL = wave * 16 + half * 8 + e;
      int d  = j * 16 + l15;
      ob[(long)(qBase + mL) * 256 + head * 32 + d] = accd[j][e];
    }
}

// ---------------------------------------------------------------------------
// Depthwise 3x3 pad1 stride1 (+ optional BN scale, bias, GELU always).
// ---------------------------------------------------------------------------
__global__ void k_dw3(const float* __restrict__ in, const float* __restrict__ w,
                      const float* __restrict__ scale, float scaleMul,
                      const float* __restrict__ bias, float* __restrict__ out,
                      int C, int H, int W, int total)
{
  int i = blockIdx.x * 256 + threadIdx.x;
  if (i >= total) return;
  int HW = H * W;
  int p = i % HW; int tmp = i / HW;
  int c = tmp % C; int b = tmp / C;
  int y = p / W, x = p - y * W;
  const float* ib = in + ((long)b * C + c) * HW;
  const float* wc = w + c * 9;
  float s = 0.f;
#pragma unroll
  for (int r = 0; r < 3; r++) {
    int iy = y - 1 + r;
    if (iy < 0 || iy >= H) continue;
#pragma unroll
    for (int t = 0; t < 3; t++) {
      int ix = x - 1 + t;
      if (ix < 0 || ix >= W) continue;
      s += ib[iy * W + ix] * wc[r * 3 + t];
    }
  }
  float sc = scale ? scale[c] * scaleMul : 1.f;
  s = s * sc + bias[c];
  out[i] = gelu_f(s);
}

// maxpool 3x3 stride2 pad1 (NCHW), out H/2 x W/2
__global__ void k_pool(const float* __restrict__ in, float* __restrict__ out,
                       int C, int H, int W, int total)
{
  int i = blockIdx.x * 256 + threadIdx.x;
  if (i >= total) return;
  int HO = H >> 1, WO = W >> 1;
  int p = i % (HO * WO); int tmp = i / (HO * WO);
  int c = tmp % C; int b = tmp / C;
  int y = p / WO, x = p - y * WO;
  const float* ib = in + ((long)b * C + c) * H * W;
  float m = -3.4e38f;
  for (int r = 0; r < 3; r++) {
    int iy = 2 * y - 1 + r;
    if (iy < 0 || iy >= H) continue;
    for (int t = 0; t < 3; t++) {
      int ix = 2 * x - 1 + t;
      if (ix < 0 || ix >= W) continue;
      m = fmaxf(m, ib[iy * W + ix]);
    }
  }
  out[i] = m;
}

// LayerNorm over last dim C; one wave per token.
__global__ __launch_bounds__(32)
void k_ln(const float* __restrict__ in, const float* __restrict__ g,
          const float* __restrict__ b, float* __restrict__ out, int C)
{
  long t = blockIdx.x;
  int lane = threadIdx.x;
  const float* p = in + t * C;
  float* o = out + t * C;
  float s = 0.f, s2 = 0.f;
  for (int c = lane; c < C; c += 32) { float v = p[c]; s += v; s2 += v * v; }
#pragma unroll
  for (int ofs = 16; ofs > 0; ofs >>= 1) {
    s += __shfl_xor(s, ofs, 32);
    s2 += __shfl_xor(s2, ofs, 32);
  }
  float mean = s / C, var = s2 / C - mean * mean;
  float r = rsqrtf(var + 1e-5f);
  for (int c = lane; c < C; c += 32) o[c] = (p[c] - mean) * r * g[c] + b[c];
}

// (B,C,P) -> (B,P,C)
__global__ void k_nchw2tok(const float* __restrict__ in, float* __restrict__ out,
                           int C, int P, int total)
{
  int i = blockIdx.x * 256 + threadIdx.x;
  if (i >= total) return;
  int c = i % C; int tmp = i / C;
  int p = tmp % P; int b = tmp / P;
  out[i] = in[((long)b * C + c) * P + p];
}

// (B,P,C) tokens -> (B,C,P) NCHW, optional residual add (NCHW)
__global__ void k_tok2nchw(const float* __restrict__ t, const float* __restrict__ res,
                           float* __restrict__ out, int C, int P, int total)
{
  int i = blockIdx.x * 256 + threadIdx.x;
  if (i >= total) return;
  int p = i % P; int tmp = i / P;
  int c = tmp % C; int b = tmp / C;
  float v = t[((long)b * P + p) * C + c];
  if (res) v += res[i];
  out[i] = v;
}

__global__ void k_copy(const float* __restrict__ in, float* __restrict__ out, int total)
{
  int i = blockIdx.x * 256 + threadIdx.x;
  if (i < total) out[i] = in[i];
}

// ---------------------------------------------------------------------------
// Host orchestration
// ---------------------------------------------------------------------------
static inline int ceilDiv(int a, int b) { return (a + b - 1) / b; }

static void gemm(hipStream_t st,
                 const float* A, long aB, int lda,
                 const float* Bm, long bB, int ldb,
                 float* C, long cB,
                 int M, int N, int K, int batch,
                 int convGather, int ks, int stride, int pad, int inW, int inH, int outW,
                 const float* rowScale, const float* rowBias, float scaleMul,
                 const float* colBias, const float* residual, long resB, int geluF)
{
  dim3 g(ceilDiv(N, 128), ceilDiv(M, 64), batch), blk(128);
  k_gemm<<<g, blk, 0, st>>>(A, aB, lda, Bm, bB, ldb, C, cB, M, N, K,
                            convGather, ks, stride, pad, inW, inH, outW,
                            rowScale, rowBias, scaleMul, colBias, residual, resB, geluF);
}

extern "C" void kernel_launch(void* const* d_in, const int* in_sizes, int n_in,
                              void* d_out, int out_size, void* d_ws, size_t ws_size,
                              hipStream_t stream)
{
  (void)in_sizes; (void)n_in; (void)out_size; (void)ws_size;
  int ip = 0;
  auto IN = [&]() -> const float* { return (const float*)d_in[ip++]; };

  const float* x = IN();
  const float *s0w0 = IN(), *s0g0 = IN(), *s0b0 = IN();
  const float *s0w1 = IN(), *s0g1 = IN(), *s0b1 = IN();

  struct MB { const float *ew,*g1,*b1,*dw,*g2,*b2,*pw,*g3,*b3,*dsw; };
  auto rdMB = [&](bool ds) {
    MB m;
    m.ew = IN(); m.g1 = IN(); m.b1 = IN();
    m.dw = IN(); m.g2 = IN(); m.b2 = IN();
    m.pw = IN(); m.g3 = IN(); m.b3 = IN();
    m.dsw = ds ? IN() : nullptr;
    return m;
  };
  MB s1p[2]; s1p[0] = rdMB(true);  s1p[1] = rdMB(false);
  MB s2p[3]; s2p[0] = rdMB(true);  s2p[1] = rdMB(false); s2p[2] = rdMB(false);

  struct TF { const float *n1g,*n1b,*qkv,*rel,*pw,*pb,*n2g,*n2b,*f1w,*f1b,*dww,*dwb,*f2w,*f2b,*dsw; };
  auto rdTF = [&](bool ds) {
    TF t;
    t.n1g = IN(); t.n1b = IN(); t.qkv = IN(); t.rel = IN();
    t.pw = IN(); t.pb = IN(); t.n2g = IN(); t.n2b = IN();
    t.f1w = IN(); t.f1b = IN(); t.dww = IN(); t.dwb = IN();
    t.f2w = IN(); t.f2b = IN();
    t.dsw = ds ? IN() : nullptr;
    return t;
  };
  TF s3p[5]; s3p[0] = rdTF(true);
  for (int i = 1; i < 5; i++) s3p[i] = rdTF(false);
  TF s4p[2]; s4p[0] = rdTF(true); s4p[1] = rdTF(false);

  // workspace slots (floats)
  float* ws = (float*)d_ws;
  const long BIG = 16L * 256 * 64 * 64;   // 16,777,216
  const long SML = 16L * 256 * 1024;      //  4,194,304
  float* sx = ws;
  float* sa = sx + BIG;
  float* sb = sa + BIG;
  float* sc = sb + BIG;
  float* sd = sc + BIG;
  float* se = sd + SML;
  float* sf = se + SML;
  float* sg = sf + SML;

  const float BNS = BNS_CONST;
  const int B = 16;

  // ---------------- stem (im2col WMMA GEMMs) ----------------
  // conv1: 3->64, 3x3 s2 p1, 256->128
  gemm(stream, s0w0, 0, 27, x, 3L * 256 * 256, 0, sa, 64L * 128 * 128,
       64, 128 * 128, 27, B, 1, 3, 2, 1, 256, 256, 128,
       s0g0, s0b0, BNS, nullptr, nullptr, 0, 1);
  // conv2: 64->64, 3x3 s1 p1 @128
  gemm(stream, s0w1, 0, 576, sa, 64L * 128 * 128, 0, sx, 64L * 128 * 128,
       64, 128 * 128, 576, B, 1, 3, 1, 1, 128, 128, 128,
       s0g1, s0b1, BNS, nullptr, nullptr, 0, 1);

  auto runMB = [&](const MB& m, int inC, int outC, int H, int W, bool ds) {
    int hid = inC * 4;
    int Ho = ds ? H / 2 : H, Wo = ds ? W / 2 : W;
    long inB = (long)inC * H * W;
    long hidB = (long)hid * Ho * Wo;
    long outB = (long)outC * Ho * Wo;
    // expand 1x1 (+BN+GELU)
    if (ds)
      gemm(stream, m.ew, 0, inC, sx, inB, 0, sa, hidB,
           hid, Ho * Wo, inC, B, 1, 1, 2, 0, W, H, Wo,
           m.g1, m.b1, BNS, nullptr, nullptr, 0, 1);
    else
      gemm(stream, m.ew, 0, inC, sx, inB, H * W, sa, hidB,
           hid, H * W, inC, B, 0, 0, 0, 0, 0, 0, 0,
           m.g1, m.b1, BNS, nullptr, nullptr, 0, 1);
    // depthwise 3x3 (+BN+GELU)
    {
      long tot = (long)B * hid * Ho * Wo;
      k_dw3<<<ceilDiv((int)tot, 256), 256, 0, stream>>>(
          sa, m.dw, m.g2, BNS, m.b2, sc, hid, Ho, Wo, (int)tot);
    }
    if (ds) {
      // project (+BN) -> sb
      gemm(stream, m.pw, 0, hid, sc, hidB, Ho * Wo, sb, outB,
           outC, Ho * Wo, hid, B, 0, 0, 0, 0, 0, 0, 0,
           m.g3, m.b3, BNS, nullptr, nullptr, 0, 0);
      // pool x -> sd
      {
        long tot = (long)B * inC * Ho * Wo;
        k_pool<<<ceilDiv((int)tot, 256), 256, 0, stream>>>(sx, sd, inC, H, W, (int)tot);
      }
      // ds_proj(pool) + h -> sx
      gemm(stream, m.dsw, 0, inC, sd, (long)inC * Ho * Wo, Ho * Wo, sx, outB,
           outC, Ho * Wo, inC, B, 0, 0, 0, 0, 0, 0, 0,
           nullptr, nullptr, 1.f, nullptr, sb, outB, 0);
    } else {
      // project (+BN) + residual x -> sx (per-element in-place safe)
      gemm(stream, m.pw, 0, hid, sc, hidB, Ho * Wo, sx, outB,
           outC, Ho * Wo, hid, B, 0, 0, 0, 0, 0, 0, 0,
           m.g3, m.b3, BNS, nullptr, sx, outB, 0);
    }
  };

  auto runTF = [&](const TF& t, int inC, int outC, int ih, int iw, bool ds) {
    int N = ih * iw;
    int hid = inC * 4;
    const float* x1;
    if (ds) {
      long tot = (long)B * inC * N;
      k_pool<<<ceilDiv((int)tot, 256), 256, 0, stream>>>(sx, sb, inC, 2 * ih, 2 * iw, (int)tot);
      x1 = sb;
    } else {
      x1 = sx;
    }
    // tokens of x1 -> sd ; LN1 -> se
    {
      long tot = (long)B * inC * N;
      k_nchw2tok<<<ceilDiv((int)tot, 256), 256, 0, stream>>>(x1, sd, inC, N, (int)tot);
    }
    k_ln<<<B * N, 32, 0, stream>>>(sd, t.n1g, t.n1b, se, inC);
    // qkv -> sf
    gemm(stream, se, (long)N * inC, inC, t.qkv, 0, 768, sf, (long)N * 768,
         N, 768, inC, B, 0, 0, 0, 0, 0, 0, 0,
         nullptr, nullptr, 1.f, nullptr, nullptr, 0, 0);
    // fused attention -> sg (B,N,256)
    {
      size_t ldsB = (size_t)(64 + N) * ALD * 2 + (size_t)32 * (N + 8) * 2 +
                    (size_t)64 * N * 4;
      dim3 g(N / 64, 8, B);
      k_attn<<<g, 128, ldsB, stream>>>(sf, t.rel, sg, N, ih, iw);
    }
    // proj (+bias) -> se
    gemm(stream, sg, (long)N * 256, 256, t.pw, 0, outC, se, (long)N * outC,
         N, outC, 256, B, 0, 0, 0, 0, 0, 0, 0,
         nullptr, nullptr, 1.f, t.pb, nullptr, 0, 0);
    // x2 branch
    const float* x2;
    if (ds) {
      gemm(stream, t.dsw, 0, inC, sb, (long)inC * N, N, sa, (long)outC * N,
           outC, N, inC, B, 0, 0, 0, 0, 0, 0, 0,
           nullptr, nullptr, 1.f, nullptr, nullptr, 0, 0);
      x2 = sa;
    } else {
      x2 = sx;
    }
    // x3 = nchw(t_proj) + x2 -> sb
    {
      long tot = (long)B * outC * N;
      k_tok2nchw<<<ceilDiv((int)tot, 256), 256, 0, stream>>>(se, x2, sb, outC, N, (int)tot);
    }
    // tokens(x3) -> sd ; LN2 -> se
    {
      long tot = (long)B * outC * N;
      k_nchw2tok<<<ceilDiv((int)tot, 256), 256, 0, stream>>>(sb, sd, outC, N, (int)tot);
    }
    k_ln<<<B * N, 32, 0, stream>>>(sd, t.n2g, t.n2b, se, outC);
    // fc1 (+bias) -> sc (B,N,hid)
    gemm(stream, se, (long)N * outC, outC, t.f1w, 0, hid, sc, (long)N * hid,
         N, hid, outC, B, 0, 0, 0, 0, 0, 0, 0,
         nullptr, nullptr, 1.f, t.f1b, nullptr, 0, 0);
    // -> NCHW (sa), depthwise 3x3 + bias + GELU (sc), -> tokens (sa)
    {
      long tot = (long)B * hid * N;
      k_tok2nchw<<<ceilDiv((int)tot, 256), 256, 0, stream>>>(sc, nullptr, sa, hid, N, (int)tot);
      k_dw3<<<ceilDiv((int)tot, 256), 256, 0, stream>>>(sa, t.dww, nullptr, 1.f, t.dwb,
                                                        sc, hid, ih, iw, (int)tot);
      k_nchw2tok<<<ceilDiv((int)tot, 256), 256, 0, stream>>>(sc, sa, hid, N, (int)tot);
    }
    // fc2 (+bias) -> se
    gemm(stream, sa, (long)N * hid, hid, t.f2w, 0, outC, se, (long)N * outC,
         N, outC, hid, B, 0, 0, 0, 0, 0, 0, 0,
         nullptr, nullptr, 1.f, t.f2b, nullptr, 0, 0);
    // out = nchw(t4) + x3 -> sx
    {
      long tot = (long)B * outC * N;
      k_tok2nchw<<<ceilDiv((int)tot, 256), 256, 0, stream>>>(se, sb, sx, outC, N, (int)tot);
    }
  };

  float* out = (float*)d_out;

  // ---------------- s1 ----------------
  runMB(s1p[0], 64, 64, 128, 128, true);
  runMB(s1p[1], 64, 64, 64, 64, false);
  {
    int tot = 16 * 64 * 64 * 64;
    k_copy<<<ceilDiv(tot, 256), 256, 0, stream>>>(sx, out, tot);
  }
  // ---------------- s2 ----------------
  runMB(s2p[0], 64, 128, 64, 64, true);
  runMB(s2p[1], 128, 128, 32, 32, false);
  runMB(s2p[2], 128, 128, 32, 32, false);
  {
    int tot = 16 * 128 * 32 * 32;
    k_copy<<<ceilDiv(tot, 256), 256, 0, stream>>>(sx, out + 4194304, tot);
  }
  // ---------------- s3 ----------------
  runTF(s3p[0], 128, 256, 16, 16, true);
  for (int i = 1; i < 5; i++) runTF(s3p[i], 256, 256, 16, 16, false);
  {
    int tot = 16 * 256 * 16 * 16;
    k_copy<<<ceilDiv(tot, 256), 256, 0, stream>>>(sx, out + 6291456, tot);
  }
  // ---------------- s4 ----------------
  runTF(s4p[0], 256, 512, 8, 8, true);
  runTF(s4p[1], 512, 512, 8, 8, false);
  {
    int tot = 16 * 512 * 8 * 8;
    k_copy<<<ceilDiv(tot, 256), 256, 0, stream>>>(sx, out + 7340032, tot);
  }
}
